// EGNN_28681791603326
// MI455X (gfx1250) — matrix operand order repaired and checked by reading, compiled, not run on hardware
//
#include <hip/hip_runtime.h>
#include <hip/hip_bf16.h>

typedef unsigned int u32;
typedef __attribute__((ext_vector_type(16))) __bf16 v16bf;
typedef __attribute__((ext_vector_type(2)))  __bf16 v2bf;
typedef __attribute__((ext_vector_type(8)))  float  v8f;

#define HDIM 64

// ---------- scalar helpers ----------
// float -> bf16 via compiler fptrunc (RNE); backend selects native cvt ops.
static __device__ __forceinline__ u32 packbf2(float lo, float hi) {
  v2bf v;
  v.x = (__bf16)lo;
  v.y = (__bf16)hi;
  return __builtin_bit_cast(u32, v);
}
static __device__ __forceinline__ unsigned short f2bf(float x) {
  return __builtin_bit_cast(unsigned short, (__bf16)x);
}
static __device__ __forceinline__ float bf2f(unsigned short h) {
  return __builtin_bit_cast(float, ((u32)h) << 16);
}
// fast sigmoid: v_exp_f32 + v_add + v_rcp_f32
static __device__ __forceinline__ float sigmoidf_(float x) {
  return __builtin_amdgcn_rcpf(1.0f + __expf(-x));
}
static __device__ __forceinline__ float siluf_(float x) { return x * sigmoidf_(x); }

// order-preserving float<->u32 encoding for atomicMax
static __device__ __forceinline__ u32 encf(float x) {
  u32 u = __builtin_bit_cast(u32, x);
  return (u & 0x80000000u) ? ~u : (u | 0x80000000u);
}
static __device__ __forceinline__ float decf(u32 e) {
  u32 u = (e & 0x80000000u) ? (e & 0x7FFFFFFFu) : ~e;
  return __builtin_bit_cast(float, u);
}

// intra-wave LDS ordering: LDS is in-order per wave; just stop compiler reordering
static __device__ __forceinline__ void wave_lds_fence() {
  __builtin_amdgcn_fence(__ATOMIC_ACQ_REL, "wavefront");
  __builtin_amdgcn_wave_barrier();
}

// ---------- fragment helpers ----------
static __device__ __forceinline__ v16bf mk_bf(uint4 a, uint4 b) {
  union { u32 u[8]; v16bf v; } x;
  x.u[0] = a.x; x.u[1] = a.y; x.u[2] = a.z; x.u[3] = a.w;
  x.u[4] = b.x; x.u[5] = b.y; x.u[6] = b.z; x.u[7] = b.w;
  return x.v;
}

static __device__ __forceinline__ v8f wmma_bf(v16bf a, v16bf b, v8f c) {
  return __builtin_amdgcn_wmma_f32_16x16x32_bf16(false, a, false, b, (short)0, c, false, false);
}

// A fragment (16x32 bf16): lane group lg = lane/16, row = lane%16 (caller passes row ptr).
static __device__ __forceinline__ v16bf load_a_bf(const unsigned short* rowp, int kb, int lg) {
  const uint4* q = (const uint4*)(rowp + kb * 32);
  return mk_bf(q[lg], q[2 + lg]);
}

// weight (B) fragment from a 256-dword-per-fragment array (global or LDS)
static __device__ __forceinline__ v16bf load_wf(const u32* frags, int fi, int lane) {
  const uint4* q = (const uint4*)(frags + fi * 256 + lane * 8);
  return mk_bf(q[0], q[1]);
}

// ---------- weight packing ----------
__global__ void pack_b_kernel(const float* __restrict__ W, int ld, int nbCols16,
                              u32* __restrict__ dst) {
  int t   = blockIdx.x * 256 + threadIdx.x;
  int fi  = t >> 8;
  int q   = t & 255;
  int lane = q >> 3, i = q & 7;
  int kb = fi / nbCols16, nb = fi % nbCols16;
  int lg = lane >> 4, n = lane & 15;
  int k   = kb * 32 + 8 * lg + 2 * i + 8 * (i >> 2);
  int col = nb * 16 + n;
  float v0 = W[(size_t)k * ld + col];
  float v1 = W[(size_t)(k + 1) * ld + col];
  dst[t] = packbf2(v0, v1);
}

__global__ void pack_row_kernel(const float* __restrict__ s, unsigned short* __restrict__ d) {
  d[threadIdx.x] = f2bf(s[threadIdx.x]);
}

__global__ void zero_kernel(u32* __restrict__ p, int n) {
  int i = blockIdx.x * 256 + threadIdx.x;
  if (i < n) p[i] = 0;
}

// ---------- input MLP: f = silu(x @ W_in + b_in) ----------
__global__ void input_kernel(const float* __restrict__ x, const u32* __restrict__ wf,
                             const float* __restrict__ b,
                             float* __restrict__ f, unsigned short* __restrict__ fbf,
                             int nTiles, int N_) {
  int wv = threadIdx.x >> 5, lane = threadIdx.x & 31;
  int tile = blockIdx.x * 8 + wv;
  if (tile >= nTiles) return;
  int lg = lane >> 4, n = lane & 15;
  int rowA = tile * 16 + n; if (rowA >= N_) rowA = N_ - 1;
  const float* rp = x + (size_t)rowA * HDIM;

  // preload all 8 weight fragments (distinct regs -> loads pipeline)
  v16bf wr[8];
#pragma unroll
  for (int i = 0; i < 8; ++i) wr[i] = load_wf(wf, i, lane);

  v16bf a[2];
#pragma unroll
  for (int kb = 0; kb < 2; ++kb) {
    const float* p = rp + kb * 32;
    float4 c0 = *(const float4*)(p + 8 * lg);
    float4 c1 = *(const float4*)(p + 8 * lg + 4);
    float4 c2 = *(const float4*)(p + 16 + 8 * lg);
    float4 c3 = *(const float4*)(p + 16 + 8 * lg + 4);
    uint4 qa = make_uint4(packbf2(c0.x, c0.y), packbf2(c0.z, c0.w),
                          packbf2(c1.x, c1.y), packbf2(c1.z, c1.w));
    uint4 qb = make_uint4(packbf2(c2.x, c2.y), packbf2(c2.z, c2.w),
                          packbf2(c3.x, c3.y), packbf2(c3.z, c3.w));
    a[kb] = mk_bf(qa, qb);
  }

  v8f acc[4];
#pragma unroll
  for (int nb = 0; nb < 4; ++nb) {
    float bb = b[nb * 16 + n];
#pragma unroll
    for (int r = 0; r < 8; ++r) acc[nb][r] = bb;
    acc[nb] = wmma_bf(a[0], wr[0 * 4 + nb], acc[nb]);
    acc[nb] = wmma_bf(a[1], wr[1 * 4 + nb], acc[nb]);
  }

  float*          pf = f   + (size_t)(tile * 16 + 8 * lg) * HDIM + n;
  unsigned short* pb = fbf + (size_t)(tile * 16 + 8 * lg) * HDIM + n;
  bool full = (tile * 16 + 16) <= N_;
  if (full) {
#pragma unroll
    for (int r = 0; r < 8; ++r)
#pragma unroll
      for (int nb = 0; nb < 4; ++nb) {
        float v = siluf_(acc[nb][r]);
        pf[r * HDIM + nb * 16] = v;
        pb[r * HDIM + nb * 16] = f2bf(v);
      }
  } else {
#pragma unroll
    for (int r = 0; r < 8; ++r) {
      if (tile * 16 + 8 * lg + r < N_)
#pragma unroll
        for (int nb = 0; nb < 4; ++nb) {
          float v = siluf_(acc[nb][r]);
          pf[r * HDIM + nb * 16] = v;
          pb[r * HDIM + nb * 16] = f2bf(v);
        }
    }
  }
}

// ---------- edge message kernel (persistent; weights cached in LDS) ----------
// wfr: 24 contiguous fragments = W1 (16) followed by W2 (8)
__global__ void edge_kernel(const unsigned short* __restrict__ fbf,
                            const float* __restrict__ w,
                            const int* __restrict__ src, const int* __restrict__ dst,
                            const u32* __restrict__ wfr,
                            const unsigned short* __restrict__ w128,
                            const float* __restrict__ b1, const float* __restrict__ b2,
                            float* __restrict__ msum, int nTiles, int E_) {
  __shared__ __align__(16) unsigned short hbuf[8][16 * HDIM];
  __shared__ __align__(16) u32 wlds[24 * 256];
  for (int i = threadIdx.x; i < 24 * 256; i += 256) wlds[i] = wfr[i];
  __syncthreads();

  int wv = threadIdx.x >> 5, lane = threadIdx.x & 31;
  int lg = lane >> 4, n = lane & 15;
  int waveId = blockIdx.x * 8 + wv;
  int nWaves = gridDim.x * 8;
  unsigned short* hb = hbuf[wv];

  // per-lane constants hoisted out of the tile loop
  float b1v[4], wlv[4], b2v[4];
#pragma unroll
  for (int nb = 0; nb < 4; ++nb) {
    b1v[nb] = b1[nb * 16 + n];
    wlv[nb] = bf2f(w128[nb * 16 + n]);
    b2v[nb] = b2[nb * 16 + n];
  }

  for (int tile = waveId; tile < nTiles; tile += nWaves) {
    int ebase = tile * 16;

    {  // prefetch next tile's edge stream
      int pre = ebase + nWaves * 16;
      if (lane == 0 && pre < E_) {
        __builtin_prefetch(src + pre, 0, 1);
        __builtin_prefetch(dst + pre, 0, 1);
        __builtin_prefetch(w + pre, 0, 1);
      }
    }

    int eA = ebase + n; if (eA >= E_) eA = E_ - 1;
    int si = src[eA], di = dst[eA];
    float wA = w[eA];
    float sqA = wA * wA;
    v16bf as0 = load_a_bf(fbf + (size_t)si * HDIM, 0, lg);
    v16bf as1 = load_a_bf(fbf + (size_t)si * HDIM, 1, lg);
    v16bf ad0 = load_a_bf(fbf + (size_t)di * HDIM, 0, lg);
    v16bf ad1 = load_a_bf(fbf + (size_t)di * HDIM, 1, lg);

    float sq[8]; int dv[8];
#pragma unroll
    for (int r = 0; r < 8; ++r) {
      sq[r] = __shfl(sqA, 8 * lg + r, 32);
      dv[r] = __shfl(di, 8 * lg + r, 32);
    }

    wave_lds_fence();   // prior-iteration LDS reads ordered before these stores

    // --- GEMM 1 + silu -> LDS (bf16) ---
#pragma unroll
    for (int nb = 0; nb < 4; ++nb) {
      v8f acc;
#pragma unroll
      for (int r = 0; r < 8; ++r) acc[r] = b1v[nb] + sq[r] * wlv[nb];
      acc = wmma_bf(as0, load_wf(wlds, 0 * 4 + nb, lane), acc);
      acc = wmma_bf(as1, load_wf(wlds, 1 * 4 + nb, lane), acc);
      acc = wmma_bf(ad0, load_wf(wlds, 2 * 4 + nb, lane), acc);
      acc = wmma_bf(ad1, load_wf(wlds, 3 * 4 + nb, lane), acc);
      unsigned short* ph = hb + (8 * lg) * HDIM + nb * 16 + n;
#pragma unroll
      for (int r = 0; r < 8; ++r) ph[r * HDIM] = f2bf(siluf_(acc[r]));
    }
    wave_lds_fence();   // stores visible to this wave's cross-lane loads

    // --- GEMM 2 ---
    v16bf h0 = load_a_bf(hb + n * HDIM, 0, lg);
    v16bf h1 = load_a_bf(hb + n * HDIM, 1, lg);
    v8f acc2[4];
#pragma unroll
    for (int nb = 0; nb < 4; ++nb) {
#pragma unroll
      for (int r = 0; r < 8; ++r) acc2[nb][r] = b2v[nb];
      acc2[nb] = wmma_bf(h0, load_wf(wlds, 16 + 0 * 4 + nb, lane), acc2[nb]);
      acc2[nb] = wmma_bf(h1, load_wf(wlds, 16 + 1 * 4 + nb, lane), acc2[nb]);
    }

    // --- gate + scatter ---
    if ((ebase + 16) <= E_) {
#pragma unroll
      for (int r = 0; r < 8; ++r) {
        float* pm = msum + (size_t)dv[r] * HDIM + n;
#pragma unroll
        for (int nb = 0; nb < 4; ++nb) {
          float msg = siluf_(acc2[nb][r]);
          float m = msg * sigmoidf_(msg);
          atomicAdd(pm + nb * 16, m);
        }
      }
    } else {
#pragma unroll
      for (int r = 0; r < 8; ++r) {
        if (ebase + 8 * lg + r < E_) {
          float* pm = msum + (size_t)dv[r] * HDIM + n;
#pragma unroll
          for (int nb = 0; nb < 4; ++nb) {
            float msg = siluf_(acc2[nb][r]);
            float m = msg * sigmoidf_(msg);
            atomicAdd(pm + nb * 16, m);
          }
        }
      }
    }
  }
}

// ---------- node update: f += silu((msum+f)@U1+b1)@U2+b2 (persistent) ----------
// wfr: 16 contiguous fragments = U1 (8) followed by U2 (8)
__global__ void update_kernel(const float* __restrict__ msum,
                              const u32* __restrict__ wfr,
                              const float* __restrict__ b1, const float* __restrict__ b2,
                              float* __restrict__ f, unsigned short* __restrict__ fbf,
                              int nTiles, int N_) {
  __shared__ __align__(16) unsigned short hbuf[8][16 * HDIM];
  __shared__ __align__(16) u32 wlds[16 * 256];
  for (int i = threadIdx.x; i < 16 * 256; i += 256) wlds[i] = wfr[i];
  __syncthreads();

  int wv = threadIdx.x >> 5, lane = threadIdx.x & 31;
  int lg = lane >> 4, n = lane & 15;
  int waveId = blockIdx.x * 8 + wv;
  int nWaves = gridDim.x * 8;
  unsigned short* hb = hbuf[wv];

  float b1v[4], b2v[4];
#pragma unroll
  for (int nb = 0; nb < 4; ++nb) {
    b1v[nb] = b1[nb * 16 + n];
    b2v[nb] = b2[nb * 16 + n];
  }

  for (int tile = waveId; tile < nTiles; tile += nWaves) {
    int nbase = tile * 16;
    bool full = (nbase + 16) <= N_;

    wave_lds_fence();
    // stage u = msum + f into LDS as bf16
    if (full) {
      const float4* pm = (const float4*)(msum + (size_t)nbase * HDIM);
      const float4* pfv = (const float4*)(f + (size_t)nbase * HDIM);
      uint2* ph = (uint2*)hb;
#pragma unroll
      for (int i = 0; i < 8; ++i) {
        int idx4 = i * 32 + lane;
        float4 a = pm[idx4];
        float4 c = pfv[idx4];
        uint2 v;
        v.x = packbf2(a.x + c.x, a.y + c.y);
        v.y = packbf2(a.z + c.z, a.w + c.w);
        ph[idx4] = v;
      }
    } else {
#pragma unroll
      for (int i = 0; i < 32; ++i) {
        int t = i * 32 + lane;
        int row = t >> 6, col = t & 63;
        int node = nbase + row; if (node >= N_) node = N_ - 1;
        size_t idx = (size_t)node * HDIM + col;
        hb[t] = f2bf(msum[idx] + f[idx]);
      }
    }
    wave_lds_fence();

    v16bf a0 = load_a_bf(hb + n * HDIM, 0, lg);
    v16bf a1 = load_a_bf(hb + n * HDIM, 1, lg);
#pragma unroll
    for (int nb = 0; nb < 4; ++nb) {
      v8f acc;
#pragma unroll
      for (int r = 0; r < 8; ++r) acc[r] = b1v[nb];
      acc = wmma_bf(a0, load_wf(wlds, 0 * 4 + nb, lane), acc);
      acc = wmma_bf(a1, load_wf(wlds, 1 * 4 + nb, lane), acc);
      unsigned short* ph = hb + (8 * lg) * HDIM + nb * 16 + n;
#pragma unroll
      for (int r = 0; r < 8; ++r) ph[r * HDIM] = f2bf(siluf_(acc[r]));
    }
    wave_lds_fence();

    v16bf c0 = load_a_bf(hb + n * HDIM, 0, lg);
    v16bf c1 = load_a_bf(hb + n * HDIM, 1, lg);
    v8f acc2[4];
#pragma unroll
    for (int nb = 0; nb < 4; ++nb) {
#pragma unroll
      for (int r = 0; r < 8; ++r) acc2[nb][r] = b2v[nb];
      acc2[nb] = wmma_bf(c0, load_wf(wlds, 8 + 0 * 4 + nb, lane), acc2[nb]);
      acc2[nb] = wmma_bf(c1, load_wf(wlds, 8 + 1 * 4 + nb, lane), acc2[nb]);
    }

    float*          pf = f   + (size_t)(nbase + 8 * lg) * HDIM + n;
    unsigned short* pb = fbf + (size_t)(nbase + 8 * lg) * HDIM + n;
    if (full) {
#pragma unroll
      for (int r = 0; r < 8; ++r)
#pragma unroll
        for (int nb = 0; nb < 4; ++nb) {
          float fn = acc2[nb][r] + pf[r * HDIM + nb * 16];   // residual
          pf[r * HDIM + nb * 16] = fn;
          pb[r * HDIM + nb * 16] = f2bf(fn);
        }
    } else {
#pragma unroll
      for (int r = 0; r < 8; ++r) {
        if (nbase + 8 * lg + r < N_)
#pragma unroll
          for (int nb = 0; nb < 4; ++nb) {
            float fn = acc2[nb][r] + pf[r * HDIM + nb * 16];
            pf[r * HDIM + nb * 16] = fn;
            pb[r * HDIM + nb * 16] = f2bf(fn);
          }
      }
    }
  }
}

// ---------- node output MLP: fout = silu(f@O1+b1)@O2+b2 (persistent) ----------
// wfr: 16 contiguous fragments = O1 (8) followed by O2 (8)
__global__ void output_kernel(const unsigned short* __restrict__ fbf,
                              const u32* __restrict__ wfr,
                              const float* __restrict__ b1, const float* __restrict__ b2,
                              float* __restrict__ fout, int nTiles, int N_) {
  __shared__ __align__(16) unsigned short hbuf[8][16 * HDIM];
  __shared__ __align__(16) u32 wlds[16 * 256];
  for (int i = threadIdx.x; i < 16 * 256; i += 256) wlds[i] = wfr[i];
  __syncthreads();

  int wv = threadIdx.x >> 5, lane = threadIdx.x & 31;
  int lg = lane >> 4, n = lane & 15;
  int waveId = blockIdx.x * 8 + wv;
  int nWaves = gridDim.x * 8;
  unsigned short* hb = hbuf[wv];

  float b1v[4], b2v[4];
#pragma unroll
  for (int nb = 0; nb < 4; ++nb) {
    b1v[nb] = b1[nb * 16 + n];
    b2v[nb] = b2[nb * 16 + n];
  }

  for (int tile = waveId; tile < nTiles; tile += nWaves) {
    int nbase = tile * 16;
    int rowA = nbase + n; if (rowA >= N_) rowA = N_ - 1;

    v16bf a0 = load_a_bf(fbf + (size_t)rowA * HDIM, 0, lg);
    v16bf a1 = load_a_bf(fbf + (size_t)rowA * HDIM, 1, lg);
    wave_lds_fence();
#pragma unroll
    for (int nb = 0; nb < 4; ++nb) {
      v8f acc;
#pragma unroll
      for (int r = 0; r < 8; ++r) acc[r] = b1v[nb];
      acc = wmma_bf(a0, load_wf(wlds, 0 * 4 + nb, lane), acc);
      acc = wmma_bf(a1, load_wf(wlds, 1 * 4 + nb, lane), acc);
      unsigned short* ph = hb + (8 * lg) * HDIM + nb * 16 + n;
#pragma unroll
      for (int r = 0; r < 8; ++r) ph[r * HDIM] = f2bf(siluf_(acc[r]));
    }
    wave_lds_fence();

    v16bf c0 = load_a_bf(hb + n * HDIM, 0, lg);
    v16bf c1 = load_a_bf(hb + n * HDIM, 1, lg);
    v8f acc2[4];
#pragma unroll
    for (int nb = 0; nb < 4; ++nb) {
#pragma unroll
      for (int r = 0; r < 8; ++r) acc2[nb][r] = b2v[nb];
      acc2[nb] = wmma_bf(c0, load_wf(wlds, 8 + 0 * 4 + nb, lane), acc2[nb]);
      acc2[nb] = wmma_bf(c1, load_wf(wlds, 8 + 1 * 4 + nb, lane), acc2[nb]);
    }

    float* po = fout + (size_t)(nbase + 8 * lg) * HDIM + n;
    if ((nbase + 16) <= N_) {
#pragma unroll
      for (int r = 0; r < 8; ++r)
#pragma unroll
        for (int nb = 0; nb < 4; ++nb)
          po[r * HDIM + nb * 16] = acc2[nb][r];
    } else {
#pragma unroll
      for (int r = 0; r < 8; ++r) {
        if (nbase + 8 * lg + r < N_)
#pragma unroll
          for (int nb = 0; nb < 4; ++nb)
            po[r * HDIM + nb * 16] = acc2[nb][r];
      }
    }
  }
}

// ---------- graph segment sum / max / count ----------
__global__ void readout_acc_kernel(const float* __restrict__ fout, const int* __restrict__ gid,
                                   float* __restrict__ gsum, u32* __restrict__ gmax,
                                   float* __restrict__ cnt, int total) {
  int t = blockIdx.x * 256 + threadIdx.x;
  if (t >= total) return;
  int node = t >> 6, col = t & 63;
  int g = gid[node];
  float v = fout[t];
  atomicAdd(&gsum[g * 64 + col], v);
  atomicMax(&gmax[g * 64 + col], encf(v));
  if (col == 0) atomicAdd(&cnt[g], 1.0f);
}

// ---------- readout MLP (tiny; scalar) ----------
__global__ void final_kernel(const float* __restrict__ gsum, const u32* __restrict__ gmax,
                             const float* __restrict__ cnt,
                             const float* __restrict__ roW1, const float* __restrict__ rob1,
                             const float* __restrict__ roW2, const float* __restrict__ rob2,
                             float* __restrict__ out) {
  __shared__ float sh[64];
  int g = blockIdx.x, n = threadIdx.x;
  float c = fmaxf(cnt[g], 1.0f);
  float rc = __builtin_amdgcn_rcpf(c);
  float acc = rob1[n];
  for (int col = 0; col < 64; ++col) {
    float s  = gsum[g * 64 + col];
    float mx = decf(gmax[g * 64 + col]);
    acc += s        * roW1[col * 64 + n];
    acc += (s * rc) * roW1[(64 + col) * 64 + n];
    acc += mx       * roW1[(128 + col) * 64 + n];
  }
  sh[n] = siluf_(acc) * roW2[n];
  __syncthreads();
  if (n == 0) {
    float s = rob2[0];
    for (int i = 0; i < 64; ++i) s += sh[i];
    out[g] = s;
  }
}

// ---------- host launcher ----------
extern "C" void kernel_launch(void* const* d_in, const int* in_sizes, int n_in,
                              void* d_out, int out_size, void* d_ws, size_t ws_size,
                              hipStream_t stream) {
  const float* x     = (const float*)d_in[0];
  const float* w     = (const float*)d_in[1];
  const int*   src   = (const int*)d_in[2];
  const int*   dst   = (const int*)d_in[3];
  const int*   gid   = (const int*)d_in[4];
  const float* W_in  = (const float*)d_in[5];
  const float* b_in  = (const float*)d_in[6];
  const float* msgW1 = (const float*)d_in[7];
  const float* msgb1 = (const float*)d_in[8];
  const float* msgW2 = (const float*)d_in[9];
  const float* msgb2 = (const float*)d_in[10];
  const float* updW1 = (const float*)d_in[11];
  const float* updb1 = (const float*)d_in[12];
  const float* updW2 = (const float*)d_in[13];
  const float* updb2 = (const float*)d_in[14];
  const float* outW1 = (const float*)d_in[15];
  const float* outb1 = (const float*)d_in[16];
  const float* outW2 = (const float*)d_in[17];
  const float* outb2 = (const float*)d_in[18];
  const float* roW1  = (const float*)d_in[19];
  const float* rob1  = (const float*)d_in[20];
  const float* roW2  = (const float*)d_in[21];
  const float* rob2  = (const float*)d_in[22];

  const int N  = in_sizes[0] / HDIM;
  const int E  = in_sizes[1];
  const int L_ = in_sizes[7] / (129 * HDIM);
  const int G_ = out_size;   // T == 1

  char* base = (char*)d_ws;
  size_t off = 0;
  auto take = [&](size_t bytes) -> char* {
    off = (off + 255) & ~(size_t)255;
    char* p = base + off;
    off += bytes;
    return p;
  };
  float*          f    = (float*)take((size_t)N * HDIM * 4);
  unsigned short* fbf  = (unsigned short*)take((size_t)N * HDIM * 2);
  float*          msum = (float*)take((size_t)N * HDIM * 4);
  float*          gsum = (float*)take((size_t)G_ * HDIM * 4);
  u32*            gmax = (u32*)take((size_t)G_ * HDIM * 4);
  float*          cnt  = (float*)take((size_t)G_ * 4);
  u32*            frags = (u32*)take((size_t)(8 + L_ * 40 + 16) * 256 * 4);
  unsigned short* w128  = (unsigned short*)take((size_t)L_ * HDIM * 2);
  (void)ws_size; (void)n_in;

  const int FR_WIN = 0;
  auto fr_m1 = [&](int l) { return 8 + l * 40; };       // 16 frags (W1) + 8 (W2) contiguous
  auto fr_m2 = [&](int l) { return 8 + l * 40 + 16; };
  auto fr_u1 = [&](int l) { return 8 + l * 40 + 24; };  // 8 (U1) + 8 (U2) contiguous
  auto fr_u2 = [&](int l) { return 8 + l * 40 + 32; };
  const int FR_O1 = 8 + L_ * 40;                        // 8 (O1) + 8 (O2) contiguous
  const int FR_O2 = FR_O1 + 8;

  // --- pack weights into WMMA B-fragment layout (bf16) ---
  pack_b_kernel<<<8, 256, 0, stream>>>(W_in, HDIM, 4, frags + (size_t)FR_WIN * 256);
  for (int l = 0; l < L_; ++l) {
    pack_b_kernel<<<16, 256, 0, stream>>>(msgW1 + (size_t)l * 129 * HDIM, HDIM, 4,
                                          frags + (size_t)fr_m1(l) * 256);
    pack_row_kernel<<<1, 64, 0, stream>>>(msgW1 + (size_t)l * 129 * HDIM + 128 * HDIM,
                                          w128 + (size_t)l * HDIM);
    pack_b_kernel<<<8, 256, 0, stream>>>(msgW2 + (size_t)l * HDIM * HDIM, HDIM, 4,
                                         frags + (size_t)fr_m2(l) * 256);
    pack_b_kernel<<<8, 256, 0, stream>>>(updW1 + (size_t)l * HDIM * HDIM, HDIM, 4,
                                         frags + (size_t)fr_u1(l) * 256);
    pack_b_kernel<<<8, 256, 0, stream>>>(updW2 + (size_t)l * HDIM * HDIM, HDIM, 4,
                                         frags + (size_t)fr_u2(l) * 256);
  }
  pack_b_kernel<<<8, 256, 0, stream>>>(outW1, HDIM, 4, frags + (size_t)FR_O1 * 256);
  pack_b_kernel<<<8, 256, 0, stream>>>(outW2, HDIM, 4, frags + (size_t)FR_O2 * 256);

  const int nTilesN = (N + 15) / 16;
  const int blkN    = (nTilesN + 7) / 8;
  const int nTilesE = (E + 15) / 16;
  const int blkE    = (nTilesE + 7) / 8;
  const int gridE   = blkE < 1024 ? blkE : 1024;   // persistent
  const int gridN   = blkN < 1024 ? blkN : 1024;   // persistent
  const int zN      = N * HDIM;
  const int zBlk    = (zN + 255) / 256;

  // --- input MLP ---
  input_kernel<<<blkN, 256, 0, stream>>>(x, frags + (size_t)FR_WIN * 256, b_in,
                                         f, fbf, nTilesN, N);

  // --- message-passing layers ---
  for (int l = 0; l < L_; ++l) {
    zero_kernel<<<zBlk, 256, 0, stream>>>((u32*)msum, zN);
    edge_kernel<<<gridE, 256, 0, stream>>>(fbf, w, src, dst,
                                           frags + (size_t)fr_m1(l) * 256,
                                           w128 + (size_t)l * HDIM,
                                           msgb1 + (size_t)l * HDIM,
                                           msgb2 + (size_t)l * HDIM,
                                           msum, nTilesE, E);
    update_kernel<<<gridN, 256, 0, stream>>>(msum,
                                             frags + (size_t)fr_u1(l) * 256,
                                             updb1 + (size_t)l * HDIM,
                                             updb2 + (size_t)l * HDIM,
                                             f, fbf, nTilesN, N);
  }

  // --- node output MLP (fout aliases msum; msum is dead now) ---
  float* fout = msum;
  output_kernel<<<gridN, 256, 0, stream>>>(fbf, frags + (size_t)FR_O1 * 256,
                                           outb1, outb2, fout, nTilesN, N);

  // --- graph readout ---
  zero_kernel<<<(G_ * HDIM + 255) / 256, 256, 0, stream>>>((u32*)gsum, G_ * HDIM);
  zero_kernel<<<(G_ * HDIM + 255) / 256, 256, 0, stream>>>(gmax, G_ * HDIM);
  zero_kernel<<<1, 256, 0, stream>>>((u32*)cnt, G_);
  readout_acc_kernel<<<(N * HDIM + 255) / 256, 256, 0, stream>>>(fout, gid, gsum, gmax,
                                                                 cnt, N * HDIM);
  final_kernel<<<G_, 64, 0, stream>>>(gsum, gmax, cnt, roW1, rob1, roW2, rob2,
                                      (float*)d_out);
}
